// MultiHeadAttention_49100066128105
// MI455X (gfx1250) — matrix operand order repaired and checked by reading, not compiled, not run on hardware
//
#include <hip/hip_runtime.h>
#include <hip/hip_fp16.h>
#include <math.h>
#include <stdint.h>

typedef __attribute__((ext_vector_type(16))) _Float16 v16h;
typedef __attribute__((ext_vector_type(8)))  _Float16 v8h;
typedef __attribute__((ext_vector_type(8)))  float    v8f;
typedef __attribute__((ext_vector_type(4)))  uint32_t u32x4;
typedef __attribute__((ext_vector_type(8)))  uint32_t u32x8;

#define B_  4
#define S_  2048
#define D_  1024
#define H_  16
#define HD_ 64

// ---------------------------------------------------------------------------
// WMMA helper: D = A(16x32 f16) * B(32x16 f16) + C(16x16 f32)
// ---------------------------------------------------------------------------
__device__ __forceinline__ v8f wmma_f16(v16h a, v16h b, v8f c) {
  return __builtin_amdgcn_wmma_f32_16x16x32_f16(
      /*neg_a=*/false, a, /*neg_b=*/false, b,
      /*c_mod=*/(short)0, c, /*reuse_a=*/false, /*reuse_b=*/false);
}

// A-fragment (16x32, 16-bit): lane L(0-15) row M=L holds K in {2*kb..2*kb+15}
// with kb = (lane>>4)*8.  Two contiguous 8-half runs in a row-major LDS row.
__device__ __forceinline__ v16h make_a_frag(const _Float16* row, int kb) {
  v8h lo = *(const v8h*)(row + 2 * kb);
  v8h hi = *(const v8h*)(row + 2 * kb + 8);
  return __builtin_shufflevector(lo, hi, 0,1,2,3,4,5,6,7,8,9,10,11,12,13,14,15);
}

// LDS staging loaders: 8 contiguous elements -> 8 halves
__device__ __forceinline__ void load8_to_lds(_Float16* dst, const float* src) {
  float4 a = *(const float4*)src;
  float4 b = *(const float4*)(src + 4);
  dst[0] = (_Float16)a.x; dst[1] = (_Float16)a.y;
  dst[2] = (_Float16)a.z; dst[3] = (_Float16)a.w;
  dst[4] = (_Float16)b.x; dst[5] = (_Float16)b.y;
  dst[6] = (_Float16)b.z; dst[7] = (_Float16)b.w;
}
__device__ __forceinline__ void load8_to_lds(_Float16* dst, const _Float16* src) {
  *(v8h*)dst = *(const v8h*)src;
}

// ---------------------------------------------------------------------------
// Tensor Data Mover: DMA a [rows x 64] f16 row-major tile (contiguous rows,
// stride 64 elements) from global memory into LDS at byte offset lds_off.
// D# layout per CDNA5 ISA (2-group form,
// VADDR2/3 = NULL -> tensors up to 2D).  Tracked by TENSORcnt.
// ---------------------------------------------------------------------------
__device__ __forceinline__ void tdm_load_tile(uint32_t lds_off,
                                              const _Float16* gsrc, int rows) {
  uint64_t ga = (uint64_t)(uintptr_t)gsrc;
  u32x4 g0;
  g0[0] = 1u;                                   // count=1, user mode
  g0[1] = lds_off;                              // lds_addr (bytes)
  g0[2] = (uint32_t)ga;                         // global_addr[31:0]
  g0[3] = (uint32_t)(ga >> 32) | (2u << 30);    // global_addr[56:32] | type=2
  u32x8 g1;
  g1[0] = 1u << 16;                             // wg_mask=0, data_size=1 (2B)
  g1[1] = (uint32_t)HD_ << 16;                  // tensor_dim0[15:0] = 64
  g1[2] = (uint32_t)S_ << 16;                   // dim0 hi=0 | tensor_dim1 lo = S
  g1[3] = (uint32_t)HD_ << 16;                  // dim1 hi=0 | tile_dim0 = 64
  g1[4] = (uint32_t)rows;                       // tile_dim1 = rows, tile_dim2=0
  g1[5] = (uint32_t)HD_;                        // tensor_dim0_stride lo = 64
  g1[6] = 0u;                                   // stride hi | dim1_stride lo
  g1[7] = 0u;                                   // dim1_stride hi
  asm volatile("tensor_load_to_lds %0, %1" :: "s"(g0), "s"(g1) : "memory");
}

// ---------------------------------------------------------------------------
// GEMM: Y[M=8192, N=1024] = X[M,K=1024] @ W[K,N] + bias[N]
// WG = 256 threads (8 waves, 4x2), WG tile 128x128, wave tile 32x64
// (8 WMMAs per K-step from 2 A-frags + 4 B-frags).
// HEADMAJOR=true  -> f16 output in [B,H,S,HD] layout (QKV projections)
// HEADMAJOR=false -> f32 output row-major [M,N]      (final projection)
// ---------------------------------------------------------------------------
template <typename TX, bool HEADMAJOR>
__global__ __launch_bounds__(256)
void gemm_wmma(const TX* __restrict__ X, const float* __restrict__ W,
               const float* __restrict__ bias, void* __restrict__ outp) {
  __shared__ _Float16 As[128][32];   // A tile, row-major
  __shared__ _Float16 Bt[128][32];   // B tile, transposed: Bt[n][k]

  const int t    = threadIdx.x;
  const int lane = t & 31;
  const int wave = t >> 5;
  const int wm   = wave >> 1;        // 0..3  -> M offset wm*32
  const int wn   = wave & 1;         // 0..1  -> N offset wn*64
  const int ln   = lane & 15;
  const int hf   = lane >> 4;
  const int m0   = blockIdx.x * 128;
  const int n0   = blockIdx.y * 128;

  v8f acc[2][4];
  for (int m = 0; m < 2; ++m)
    for (int i = 0; i < 4; ++i) { v8f z = {}; acc[m][i] = z; }

  for (int kc = 0; kc < D_; kc += 32) {
    // --- stage A tile 128x32 (16 elems / thread) ---
    {
      int r = t >> 1;                  // 0..127
      int c = (t & 1) * 16;            // 0 or 16
      const TX* src = X + (size_t)(m0 + r) * D_ + kc + c;
      load8_to_lds(&As[r][c], src);
      load8_to_lds(&As[r][c + 8], src + 8);
    }
    // --- stage B tile 32x128, transposed into Bt[n][k] (16 elems / thread) ---
    {
      int kk = t >> 3;                 // 0..31
      int c0 = (t & 7) * 16;           // 0..112
      const float* src = W + (size_t)(kc + kk) * D_ + n0 + c0;
#pragma unroll
      for (int j = 0; j < 16; j += 4) {
        float4 w4 = *(const float4*)(src + j);
        Bt[c0 + j + 0][kk] = (_Float16)w4.x;
        Bt[c0 + j + 1][kk] = (_Float16)w4.y;
        Bt[c0 + j + 2][kk] = (_Float16)w4.z;
        Bt[c0 + j + 3][kk] = (_Float16)w4.w;
      }
    }
    __syncthreads();

    v16h a0 = make_a_frag(&As[wm * 32 + ln][0], hf * 8);
    v16h a1 = make_a_frag(&As[wm * 32 + 16 + ln][0], hf * 8);
#pragma unroll
    for (int tt = 0; tt < 4; ++tt) {
      // B frag: lane = column, K 0..15 (lanes 0-15) / 16..31 (lanes 16-31)
      v16h b = *(const v16h*)&Bt[wn * 64 + tt * 16 + ln][hf * 16];
      acc[0][tt] = wmma_f16(a0, b, acc[0][tt]);
      acc[1][tt] = wmma_f16(a1, b, acc[1][tt]);
    }
    __syncthreads();
  }

  // --- store: C layout = lane column, VGPR i -> row i (+8 for upper half) ---
#pragma unroll
  for (int ms = 0; ms < 2; ++ms) {
#pragma unroll
    for (int tt = 0; tt < 4; ++tt) {
      int gc = n0 + wn * 64 + tt * 16 + ln;
      float bb = bias[gc];
#pragma unroll
      for (int i = 0; i < 8; ++i) {
        int gm = m0 + wm * 32 + ms * 16 + i + 8 * hf;
        float val = acc[ms][tt][i] + bb;
        if constexpr (HEADMAJOR) {
          int bb_ = gm >> 11, ss = gm & (S_ - 1);
          int hh  = gc >> 6,  dd = gc & (HD_ - 1);
          ((_Float16*)outp)[(((size_t)(bb_ * H_ + hh) * S_ + ss) * HD_) + dd] =
              (_Float16)val;
        } else {
          ((float*)outp)[(size_t)gm * D_ + gc] = val;
        }
      }
    }
  }
}

// ---------------------------------------------------------------------------
// Flash attention: one WG (4 waves) per (b, h, 64-row q block).
// Q,K,V are f16 in [B,H,S,HD]; output f16 in [B,S,D] (heads re-interleaved).
// Q block and K chunks are DMA'd into LDS by the Tensor Data Mover.
// ---------------------------------------------------------------------------
__global__ __launch_bounds__(128)
void attn_wmma(const _Float16* __restrict__ Q, const _Float16* __restrict__ K,
               const _Float16* __restrict__ V, _Float16* __restrict__ O) {
  __shared__ _Float16 Qs[64][64];    // q block
  __shared__ _Float16 Ks[32][64];    // key chunk, row-major (rows == B^T cols)
  __shared__ _Float16 Vt[64][32];    // value chunk transposed: Vt[d][key]
  __shared__ _Float16 Ps[4][16][32]; // per-wave probability scratch

  const int t    = threadIdx.x;
  const int lane = t & 31;
  const int wave = t >> 5;           // 0..3 -> q rows wave*16..+15
  const int ln   = lane & 15;
  const int hf   = lane >> 4;
  const int q0   = blockIdx.x * 64;
  const int h    = blockIdx.y;
  const int b    = blockIdx.z;

  const size_t head = ((size_t)b * H_ + h) * (size_t)S_ * HD_;
  const _Float16* Qh = Q + head;
  const _Float16* Kh = K + head;
  const _Float16* Vh = V + head;

  const uint32_t qs_off = (uint32_t)(uintptr_t)&Qs[0][0];
  const uint32_t ks_off = (uint32_t)(uintptr_t)&Ks[0][0];

  // TDM: DMA the 64x64 Q block into LDS (wave 0 issues; barrier publishes)
  if (wave == 0) tdm_load_tile(qs_off, Qh + (size_t)q0 * HD_, 64);

  float m_run[8], l_run[8];
  v8f acc[4];
  for (int i = 0; i < 8; ++i) { m_run[i] = -INFINITY; l_run[i] = 0.f; }
  for (int i = 0; i < 4; ++i) { v8f z = {}; acc[i] = z; }

  const float scale = 0.125f;  // 1/sqrt(64)

  for (int kc = 0; kc < S_; kc += 32) {
    __syncthreads();  // all waves done with previous Ks/Vt before overwrite
    // TDM: DMA K chunk 32x64 into LDS
    if (wave == 0) tdm_load_tile(ks_off, Kh + (size_t)kc * HD_, 32);
    // stage V chunk transposed (all threads, vector global load)
    {
      int r = t >> 2;
      int c = (t & 3) * 16;
      v16h vv = *(const v16h*)(Vh + (size_t)(kc + r) * HD_ + c);
#pragma unroll
      for (int j = 0; j < 16; ++j) Vt[c + j][r] = vv[j];
    }
    if (wave == 0) __builtin_amdgcn_s_wait_tensorcnt(0);  // Q (iter 0) + K done
    __syncthreads();

    // scores: s0 = keys kc..kc+15, s1 = keys kc+16..kc+31
    v8f s0 = {}, s1 = {};
#pragma unroll
    for (int dc = 0; dc < HD_; dc += 32) {
      v16h a  = make_a_frag(&Qs[wave * 16 + ln][dc], hf * 8);
      v16h b0 = *(const v16h*)&Ks[ln][dc + hf * 16];
      v16h b1 = *(const v16h*)&Ks[16 + ln][dc + hf * 16];
      s0 = wmma_f16(a, b0, s0);
      s1 = wmma_f16(a, b1, s1);
    }

    // online softmax; C row i lives within one 16-lane half -> xor masks 1..8
#pragma unroll
    for (int i = 0; i < 8; ++i) {
      float a0 = s0[i] * scale;
      float a1 = s1[i] * scale;
      float mx = fmaxf(a0, a1);
#pragma unroll
      for (int off = 1; off < 16; off <<= 1)
        mx = fmaxf(mx, __shfl_xor(mx, off, 32));
      float m_new = fmaxf(m_run[i], mx);
      float corr  = __expf(m_run[i] - m_new);
      float p0    = __expf(a0 - m_new);
      float p1    = __expf(a1 - m_new);
      float rs    = p0 + p1;
#pragma unroll
      for (int off = 1; off < 16; off <<= 1) rs += __shfl_xor(rs, off, 32);
      l_run[i] = l_run[i] * corr + rs;
      m_run[i] = m_new;
#pragma unroll
      for (int tt = 0; tt < 4; ++tt) acc[tt][i] *= corr;
      int r = i + 8 * hf;
      Ps[wave][r][ln]      = (_Float16)p0;
      Ps[wave][r][16 + ln] = (_Float16)p1;
    }

    // O += P @ V   (per-wave Ps: same-wave DS RAW, hardware-ordered)
    v16h pa = make_a_frag(&Ps[wave][ln][0], hf * 8);
#pragma unroll
    for (int tt = 0; tt < 4; ++tt) {
      v16h vb = *(const v16h*)&Vt[tt * 16 + ln][hf * 16];
      acc[tt] = wmma_f16(pa, vb, acc[tt]);
    }
  }

  // normalize and store to [B,S,D] with head offset
#pragma unroll
  for (int tt = 0; tt < 4; ++tt) {
#pragma unroll
    for (int i = 0; i < 8; ++i) {
      int r    = i + 8 * hf;
      int qrow = q0 + wave * 16 + r;
      int d    = tt * 16 + ln;
      O[((size_t)b * S_ + qrow) * D_ + h * HD_ + d] =
          (_Float16)(acc[tt][i] / l_run[i]);
    }
  }
}

// ---------------------------------------------------------------------------
extern "C" void kernel_launch(void* const* d_in, const int* in_sizes, int n_in,
                              void* d_out, int out_size, void* d_ws, size_t ws_size,
                              hipStream_t stream) {
  const float* x  = (const float*)d_in[0];
  const float* Wq = (const float*)d_in[1];
  const float* bq = (const float*)d_in[2];
  const float* Wk = (const float*)d_in[3];
  const float* bk = (const float*)d_in[4];
  const float* Wv = (const float*)d_in[5];
  const float* bv = (const float*)d_in[6];
  const float* Wo = (const float*)d_in[7];
  const float* bo = (const float*)d_in[8];

  const size_t NELT = (size_t)B_ * S_ * D_;   // 8.39M
  _Float16* q  = (_Float16*)d_ws;
  _Float16* k  = q + NELT;
  _Float16* v  = k + NELT;
  _Float16* ao = v + NELT;                     // 64 MB total ws use

  dim3 ggrid(B_ * S_ / 128, D_ / 128);         // 64 x 8
  gemm_wmma<float, true><<<ggrid, 256, 0, stream>>>(x, Wq, bq, (void*)q);
  gemm_wmma<float, true><<<ggrid, 256, 0, stream>>>(x, Wk, bk, (void*)k);
  gemm_wmma<float, true><<<ggrid, 256, 0, stream>>>(x, Wv, bv, (void*)v);

  attn_wmma<<<dim3(S_ / 64, H_, B_), 128, 0, stream>>>(q, k, v, ao);

  gemm_wmma<_Float16, false><<<ggrid, 256, 0, stream>>>(ao, Wo, bo, d_out);
}
